// MyLSTM_87007447482641
// MI455X (gfx1250) — compile-verified, matrix operand order checked by reading, not run on hardware
//
#include <hip/hip_runtime.h>
#include <cstddef>

// ---------------- types ----------------
typedef __attribute__((ext_vector_type(16))) __bf16 v16bf;
typedef __attribute__((ext_vector_type(8)))  float  v8f;
typedef unsigned uint4e __attribute__((ext_vector_type(4)));
typedef unsigned uint2e __attribute__((ext_vector_type(2)));

union Frag { uint4e u[2]; v16bf v; };

// ---------------- constants ----------------
constexpr int T  = 4096;
constexpr int I  = 1024;
constexpr int H  = 2048;
constexpr int G  = 4 * H;   // 8192 gate rows
constexpr int O  = 512;

constexpr int NBLK = 256;           // persistent blocks for the scan
constexpr int HB   = H / NBLK;      // 8 hidden elems per block
constexpr int ROWS = 4 * HB;        // 32 gate rows per block
constexpr int WSTR = H + 8;         // padded LDS row stride (ushorts) -> kills bank conflicts
constexpr size_t SMEM_BYTES = (size_t)ROWS * WSTR * 2 + (size_t)H * 2 + ROWS * 4;

// ---------------- helpers ----------------
__device__ __forceinline__ float bf_lo(unsigned u) {
    union { unsigned i; float f; } v; v.i = u << 16; return v.f;
}
__device__ __forceinline__ float bf_hi(unsigned u) {
    union { unsigned i; float f; } v; v.i = u & 0xffff0000u; return v.f;
}
__device__ __forceinline__ unsigned short f2bf(float f) {
    union { float f; unsigned i; } v; v.f = f;
    unsigned r = (v.i + 0x7fffu + ((v.i >> 16) & 1u)) >> 16;
    return (unsigned short)r;
}
__device__ __forceinline__ float sigf(float x) { return 1.0f / (1.0f + __expf(-x)); }

// ---------------- init: zero h buffers / counter ----------------
__global__ void init_state(unsigned short* hbuf, float* hout, unsigned* counter) {
    int t = threadIdx.x;
    for (int i = t; i < 2 * H; i += blockDim.x) hbuf[i] = 0;
    for (int i = t; i < H; i += blockDim.x) hout[i] = 0.0f;
    if (t == 0) *counter = 0u;
}

// ---------------- fp32 -> bf16 ----------------
__global__ void cvt_f32_bf16(const float* __restrict__ s, unsigned short* __restrict__ d, int n) {
    for (int i = blockIdx.x * blockDim.x + threadIdx.x; i < n; i += gridDim.x * blockDim.x)
        d[i] = f2bf(s[i]);
}

// ---------------- Phase 1: x_gates = A @ W_ih^T + (b_ih + b_hh) ----------------
// block = 8 waves; wave w handles rows [m0, m0+16), all 8 N-subtiles of a 128-wide strip.
// Per K-step: burst-load A frag + all 8 B frags (18 b128 loads in flight), then run the
// 8-WMMA chain so the scheduler can overlap WMMAs with the load tail (partial loadcnt waits).
__global__ __launch_bounds__(256) void gemm_xgates(
        const unsigned short* __restrict__ Abf,   // [T][I] bf16
        const unsigned short* __restrict__ Wbf,   // [G][I] bf16
        const float* __restrict__ b_ih, const float* __restrict__ b_hh,
        float* __restrict__ xg)                   // [T][G] f32
{
    const int lane = threadIdx.x & 31;
    const int wave = threadIdx.x >> 5;
    const int r    = lane & 15;
    const int half = lane >> 4;
    const int m0   = blockIdx.y * 128 + wave * 16;
    const int n0   = blockIdx.x * 128;

    v8f acc[8] = {};
    const unsigned short* arow = Abf + (size_t)(m0 + r) * I;
    const unsigned short* brow[8];
#pragma unroll
    for (int s = 0; s < 8; ++s)
        brow[s] = Wbf + (size_t)(n0 + s * 16 + r) * I;

    for (int kk = 0; kk < I; kk += 32) {
        // Burst: issue every load of this K-step before any WMMA.
        Frag b[8];
#pragma unroll
        for (int s = 0; s < 8; ++s) {
            b[s].u[0] = *reinterpret_cast<const uint4e*>(brow[s] + kk + half * 8);
            b[s].u[1] = *reinterpret_cast<const uint4e*>(brow[s] + kk + 16 + half * 8);
        }
        Frag a;
        a.u[0] = *reinterpret_cast<const uint4e*>(arow + kk + half * 8);
        a.u[1] = *reinterpret_cast<const uint4e*>(arow + kk + 16 + half * 8);
        if (kk + 32 < I) __builtin_prefetch(arow + kk + 32, 0, 3);
#pragma unroll
        for (int s = 0; s < 8; ++s)
            acc[s] = __builtin_amdgcn_wmma_f32_16x16x32_bf16(
                false, a.v, false, b[s].v, (short)0, acc[s], false, false);
    }
#pragma unroll
    for (int s = 0; s < 8; ++s) {
        const int n = n0 + s * 16 + r;
        const float bias = b_ih[n] + b_hh[n];
#pragma unroll
        for (int v = 0; v < 8; ++v) {
            const int row = m0 + v + half * 8;
            xg[(size_t)row * G + n] = acc[s][v] + bias;
        }
    }
}

// ---------------- Phase 2: persistent LSTM scan ----------------
// 256 blocks x 256 threads. Block b owns h[8b..8b+7] and the 32 matching gate rows
// of W_hh, held in LDS (128 KB slab) for all 4096 steps. Grid barrier per step.
__global__ __launch_bounds__(256) void lstm_scan(
        const float* __restrict__ xg,             // [T][G]
        const unsigned short* __restrict__ Whh,   // [G][H] bf16
        unsigned short* hbuf,                     // [2][H] bf16 ping-pong
        float* __restrict__ hout,                 // [H] final h (f32)
        unsigned* counter)
{
    extern __shared__ char smem[];
    unsigned short* lW = (unsigned short*)smem;           // [ROWS][WSTR]
    unsigned short* lH = lW + (size_t)ROWS * WSTR;        // [H]
    float* lG = (float*)(lH + H);                         // [ROWS]

    const int tid = threadIdx.x;
    const int j0  = blockIdx.x * HB;

    // Load this block's W_hh slab into LDS once (coalesced along k).
    for (int idx = tid; idx < ROWS * H; idx += 256) {
        const int rr = idx >> 11;          // row 0..31 = gate*8 + jj
        const int k  = idx & (H - 1);
        const int gate = rr >> 3, jj = rr & 7;
        lW[rr * WSTR + k] = Whh[(size_t)(gate * H + j0 + jj) * H + k];
    }
    float c = 0.0f;                        // cell state, owned by threads 0..7
    __syncthreads();

    const int row = tid >> 3;              // 0..31
    const int seg = tid & 7;               // 0..7, 256 k-elems each

    for (int t = 0; t < T; ++t) {
        // Stage h(t-1) into LDS.
        const unsigned short* hsrc = hbuf + (t & 1) * H;
        for (int idx = tid; idx < H; idx += 256) lH[idx] = hsrc[idx];
        __syncthreads();

        // Partial dot: 256 bf16 MACs per thread, packed pair loads.
        const uint2e* w2 = reinterpret_cast<const uint2e*>(lW + row * WSTR + seg * 256);
        const uint2e* h2 = reinterpret_cast<const uint2e*>(lH + seg * 256);
        float acc = 0.0f;
#pragma unroll 4
        for (int k = 0; k < 64; ++k) {
            const uint2e wu = w2[k], hu = h2[k];
            acc = fmaf(bf_lo(wu.x), bf_lo(hu.x), acc);
            acc = fmaf(bf_hi(wu.x), bf_hi(hu.x), acc);
            acc = fmaf(bf_lo(wu.y), bf_lo(hu.y), acc);
            acc = fmaf(bf_hi(wu.y), bf_hi(hu.y), acc);
        }
        acc += __shfl_xor(acc, 4, 8);
        acc += __shfl_xor(acc, 2, 8);
        acc += __shfl_xor(acc, 1, 8);
        if (seg == 0) lG[row] = acc;
        __syncthreads();

        if (tid < HB) {
            const size_t base = (size_t)t * G;
            const int j = j0 + tid;
            const float gi = lG[tid]      + xg[base + j];
            const float gf = lG[8  + tid] + xg[base + H     + j];
            const float gg = lG[16 + tid] + xg[base + 2 * H + j];
            const float go = lG[24 + tid] + xg[base + 3 * H + j];
            c = sigf(gf) * c + sigf(gi) * tanhf(gg);
            const float hn = sigf(go) * tanhf(c);
            hbuf[((t + 1) & 1) * H + j] = f2bf(hn);
            if (t == T - 1) hout[j] = hn;
        }

        // Grid-wide barrier (monotonic counter; double-buffered h makes one sync enough).
        __threadfence();
        __syncthreads();
        if (tid == 0) {
            atomicAdd(counter, 1u);
            const unsigned target = (unsigned)(t + 1) * gridDim.x;
            while (__hip_atomic_load(counter, __ATOMIC_ACQUIRE, __HIP_MEMORY_SCOPE_AGENT) < target)
                __builtin_amdgcn_s_sleep(2);
        }
        __syncthreads();
    }
}

// ---------------- Phase 3: out = h_last @ W_lin^T + b_lin ----------------
__global__ __launch_bounds__(256) void final_proj(
        const float* __restrict__ h, const float* __restrict__ Wl,
        const float* __restrict__ bl, float* __restrict__ out)
{
    const int wave = threadIdx.x >> 5, lane = threadIdx.x & 31;
    const int o = blockIdx.x * 8 + wave;
    const float* wr = Wl + (size_t)o * H;
    float acc = 0.0f;
    for (int k = lane; k < H; k += 32) acc = fmaf(wr[k], h[k], acc);
#pragma unroll
    for (int off = 16; off >= 1; off >>= 1) acc += __shfl_xor(acc, off, 32);
    if (lane == 0) out[o] = acc + bl[o];
}

// ---------------- host ----------------
extern "C" void kernel_launch(void* const* d_in, const int* in_sizes, int n_in,
                              void* d_out, int out_size, void* d_ws, size_t ws_size,
                              hipStream_t stream) {
    const float* input_seq = (const float*)d_in[0];   // [T,I]
    const float* W_ih      = (const float*)d_in[1];   // [4H,I]
    const float* W_hh      = (const float*)d_in[2];   // [4H,H]
    const float* b_ih      = (const float*)d_in[3];   // [4H]
    const float* b_hh      = (const float*)d_in[4];   // [4H]
    const float* W_lin     = (const float*)d_in[5];   // [O,H]
    const float* b_lin     = (const float*)d_in[6];   // [O]
    float* out = (float*)d_out;

    char* ws = (char*)d_ws;
    size_t off = 0;
    auto carve = [&](size_t bytes) { char* p = ws + off; off = (off + bytes + 255) & ~(size_t)255; return p; };
    float*          xg     = (float*)         carve((size_t)T * G * 4);   // 128 MB
    unsigned short* Wih_bf = (unsigned short*)carve((size_t)G * I * 2);   // 16 MB
    unsigned short* Whh_bf = (unsigned short*)carve((size_t)G * H * 2);   // 32 MB
    unsigned short* in_bf  = (unsigned short*)carve((size_t)T * I * 2);   // 8 MB
    unsigned short* hbuf   = (unsigned short*)carve((size_t)2 * H * 2);
    float*          hout   = (float*)         carve((size_t)H * 4);
    unsigned*       ctr    = (unsigned*)      carve(256);
    (void)ws_size; (void)in_sizes; (void)n_in; (void)out_size;

    init_state<<<1, 256, 0, stream>>>(hbuf, hout, ctr);

    cvt_f32_bf16<<<2048, 256, 0, stream>>>(input_seq, in_bf,  T * I);
    cvt_f32_bf16<<<2048, 256, 0, stream>>>(W_ih,      Wih_bf, G * I);
    cvt_f32_bf16<<<4096, 256, 0, stream>>>(W_hh,      Whh_bf, G * H);

    gemm_xgates<<<dim3(G / 128, T / 128), 256, 0, stream>>>(in_bf, Wih_bf, b_ih, b_hh, xg);

    (void)hipFuncSetAttribute((const void*)lstm_scan,
                              hipFuncAttributeMaxDynamicSharedMemorySize,
                              (int)SMEM_BYTES);
    lstm_scan<<<NBLK, 256, SMEM_BYTES, stream>>>(xg, Whh_bf, hbuf, hout, ctr);

    final_proj<<<O / 8, 256, 0, stream>>>(hout, W_lin, b_lin, out);
}
